// RNN_39264591020102
// MI455X (gfx1250) — matrix operand order repaired and checked by reading, not compiled, not run on hardware
//
#include <hip/hip_runtime.h>
#include <hip/hip_bf16.h>
#include <stdint.h>

// ---------------------------------------------------------------------------
// Problem constants (match reference)
// ---------------------------------------------------------------------------
#define BB      32
#define SS      64          // S_SRC
#define TT      64          // MAX_LENGTH
#define HH      1024
#define H3      3072
#define VV      32000
#define SOS_TOK 0

typedef __attribute__((ext_vector_type(16))) __bf16 v16bf;
typedef __attribute__((ext_vector_type(8)))  float  v8f;

union TileU { v16bf v; uint4 q[2]; };

// float -> bf16, round-to-nearest-even, NaN-safe
__device__ __forceinline__ uint16_t f2bf(float f) {
    union { float f; uint32_t u; } x; x.f = f;
    uint32_t u = x.u;
    if ((u & 0x7fffffffu) > 0x7f800000u) return (uint16_t)((u >> 16) | 0x0040u);
    u += 0x7fffu + ((u >> 16) & 1u);
    return (uint16_t)(u >> 16);
}

__device__ __forceinline__ float sigmoidf_(float x) {
    return 1.0f / (1.0f + __expf(-x));
}

// ---------------------------------------------------------------------------
// WMMA tile loaders (ISA 7.12.2 layouts, wave32)
// ---------------------------------------------------------------------------
// A: 16x32 bf16 tile (MxK), source row-major with leading dim `lda` (elements).
// lanes 0-15: row=lane, K elements {0..7} then {16..23}
// lanes 16-31: row=lane-16, K elements {8..15} then {24..31}
__device__ __forceinline__ v16bf load_A_tile(const uint16_t* __restrict__ base,
                                             int lda, int lane) {
    int row  = lane & 15;
    int koff = (lane >= 16) ? 8 : 0;
    const uint16_t* p = base + (size_t)row * lda + koff;
    TileU t;
    t.q[0] = *(const uint4*)(p);         // K = koff + 0..7
    t.q[1] = *(const uint4*)(p + 16);    // K = koff + 16..23
    return t.v;
}

// B: 32x16 bf16 tile (KxN) where B(k,n) = W[n,k], W row-major [N, K] with
// leading dim `ldw`. lanes 0-15: col=lane, K {0..7} then {16..23}; lanes 16-31:
// col=lane-16, K {8..15} then {24..31} (matches A tile order).
__device__ __forceinline__ v16bf load_B_tile(const uint16_t* __restrict__ wbase,
                                             int ldw, int lane) {
    int n    = lane & 15;
    int koff = (lane >= 16) ? 8 : 0;
    const uint16_t* p = wbase + (size_t)n * ldw + koff;
    TileU t;
    t.q[0] = *(const uint4*)(p);
    t.q[1] = *(const uint4*)(p + 16);
    return t.v;
}

// ---------------------------------------------------------------------------
// Elementwise helpers
// ---------------------------------------------------------------------------
__global__ void k_f32_to_bf16(const float* __restrict__ src,
                              uint16_t* __restrict__ dst, int n) {
    int i = blockIdx.x * blockDim.x + threadIdx.x;
    if (i < n) dst[i] = f2bf(src[i]);
}

__global__ void k_zero_h(float* __restrict__ hf, uint16_t* __restrict__ hb) {
    int i = blockIdx.x * blockDim.x + threadIdx.x;
    if (i < BB * HH) { hf[i] = 0.0f; hb[i] = 0; }
}

// Encoder embedding gather -> bf16 [B, S, H]
__global__ void k_enc_embed(const int* __restrict__ tokens,
                            const float* __restrict__ emb,
                            uint16_t* __restrict__ x) {
    int i = blockIdx.x * blockDim.x + threadIdx.x;
    if (i >= BB * SS * HH) return;
    int h = i & (HH - 1);
    int s = (i >> 10) & (SS - 1);
    int b = i >> 16;
    int tok = tokens[b * SS + s];
    x[i] = f2bf(emb[(size_t)tok * HH + h]);
}

// Decoder embedding: shifted targets (SOS first), ReLU -> bf16 [B, T, H]
__global__ void k_dec_embed(const int* __restrict__ targets,
                            const float* __restrict__ emb,
                            uint16_t* __restrict__ x) {
    int i = blockIdx.x * blockDim.x + threadIdx.x;
    if (i >= BB * TT * HH) return;
    int h = i & (HH - 1);
    int t = (i >> 10) & (TT - 1);
    int b = i >> 16;
    int tok = (t == 0) ? SOS_TOK : targets[b * TT + (t - 1)];
    float v = emb[(size_t)tok * HH + h];
    x[i] = f2bf(v > 0.0f ? v : 0.0f);
}

// ---------------------------------------------------------------------------
// GRU step GEMM: one wave per 16x16 tile of [B, 3H]; computes both
//   gi = x_t * W_ih^T + b_ih   and   gh = h * W_hh^T + b_hh
// Latency-bound scan step -> maximize parallelism, not reuse.
// ---------------------------------------------------------------------------
__global__ __launch_bounds__(32)
void k_gru_gemm(const uint16_t* __restrict__ x, int ldx,
                const uint16_t* __restrict__ h,
                const uint16_t* __restrict__ w_ih,
                const uint16_t* __restrict__ w_hh,
                const float* __restrict__ b_ih,
                const float* __restrict__ b_hh,
                float* __restrict__ gi, float* __restrict__ gh) {
    int lane = threadIdx.x;
    int n0 = blockIdx.x * 16;      // 0..3071
    int m0 = blockIdx.y * 16;      // 0 or 16
    v8f ai = {0.f,0.f,0.f,0.f,0.f,0.f,0.f,0.f};
    v8f ah = {0.f,0.f,0.f,0.f,0.f,0.f,0.f,0.f};
    for (int k0 = 0; k0 < HH; k0 += 32) {
        v16bf Ax = load_A_tile(x + (size_t)m0 * ldx + k0, ldx, lane);
        v16bf Ah = load_A_tile(h + (size_t)m0 * HH + k0, HH, lane);
        v16bf Bi = load_B_tile(w_ih + (size_t)n0 * HH + k0, HH, lane);
        v16bf Bh = load_B_tile(w_hh + (size_t)n0 * HH + k0, HH, lane);
        ai = __builtin_amdgcn_wmma_f32_16x16x32_bf16(false, Ax, false, Bi,
                                                     (short)0, ai, false, false);
        ah = __builtin_amdgcn_wmma_f32_16x16x32_bf16(false, Ah, false, Bh,
                                                     (short)0, ah, false, false);
    }
    int col   = n0 + (lane & 15);
    int rbase = m0 + ((lane >> 4) << 3);
    float bi = b_ih[col], bh = b_hh[col];
#pragma unroll
    for (int e = 0; e < 8; ++e) {
        gi[(size_t)(rbase + e) * H3 + col] = ai[e] + bi;
        gh[(size_t)(rbase + e) * H3 + col] = ah[e] + bh;
    }
}

// ---------------------------------------------------------------------------
// GRU gate update: h_new = (1-z)*n + z*h ; also emits bf16 h for next step's
// WMMA A operand, optionally the decoder output slot and the final hidden.
// ---------------------------------------------------------------------------
__global__ void k_gru_gate(const float* __restrict__ gi,
                           const float* __restrict__ gh,
                           float* __restrict__ h_f32,
                           uint16_t* __restrict__ h_bf16,
                           uint16_t* __restrict__ out_bf16,  // + t*H, or null
                           float* __restrict__ h_final) {    // or null
    int i = blockIdx.x * blockDim.x + threadIdx.x;
    if (i >= BB * HH) return;
    int b = i >> 10, hh = i & (HH - 1);
    const float* gib = gi + (size_t)b * H3;
    const float* ghb = gh + (size_t)b * H3;
    float r = sigmoidf_(gib[hh]           + ghb[hh]);
    float z = sigmoidf_(gib[HH + hh]      + ghb[HH + hh]);
    float n = tanhf    (gib[2 * HH + hh]  + r * ghb[2 * HH + hh]);
    float hn = (1.0f - z) * n + z * h_f32[i];
    h_f32[i] = hn;
    uint16_t hb = f2bf(hn);
    h_bf16[i] = hb;
    if (out_bf16) out_bf16[(size_t)b * (TT * HH) + hh] = hb;
    if (h_final)  h_final[i] = hn;
}

// ---------------------------------------------------------------------------
// Output projection: [2048, 1024] x [1024, 32000] + bias -> fp32 logits.
// 4 waves / block; each wave owns a 64x64 tile (4x4 register blocking,
// 16 f32 accumulators = 128 VGPRs). Per K-step: 4 A + 4 B tile loads feed
// 16 WMMAs -> 1 b128 load per WMMA, ~4x less L2 traffic than 16x64 tiling.
// ---------------------------------------------------------------------------
__global__ __launch_bounds__(128)
void k_out_proj(const uint16_t* __restrict__ A,   // [B*T, H] bf16
                const uint16_t* __restrict__ Wb,  // [V, H]  bf16
                const float* __restrict__ bias,   // [V]
                float* __restrict__ out) {        // [B*T, V] fp32
    const v8f vzero = {0.f,0.f,0.f,0.f,0.f,0.f,0.f,0.f};
    int lane  = threadIdx.x & 31;
    int wid   = threadIdx.x >> 5;
    int mbase = blockIdx.y * 64;                 // 32 M-groups of 64 rows
    int nbase = blockIdx.x * 256 + wid * 64;     // 125 N-groups of 256 cols

    v8f acc[4][4];
#pragma unroll
    for (int i = 0; i < 4; ++i)
#pragma unroll
        for (int j = 0; j < 4; ++j) acc[i][j] = vzero;

    for (int k0 = 0; k0 < HH; k0 += 32) {
        // prefetch next K-slab of this wave's B strip (global_prefetch_b8)
        if (k0 + 32 < HH)
            __builtin_prefetch(Wb + (size_t)(nbase + (lane & 15)) * HH + k0 + 32, 0, 1);
        v16bf Av[4], Bv[4];
#pragma unroll
        for (int i = 0; i < 4; ++i)
            Av[i] = load_A_tile(A + (size_t)(mbase + i * 16) * HH + k0, HH, lane);
#pragma unroll
        for (int j = 0; j < 4; ++j)
            Bv[j] = load_B_tile(Wb + (size_t)(nbase + j * 16) * HH + k0, HH, lane);
#pragma unroll
        for (int i = 0; i < 4; ++i)
#pragma unroll
            for (int j = 0; j < 4; ++j)
                acc[i][j] = __builtin_amdgcn_wmma_f32_16x16x32_bf16(
                    false, Av[i], false, Bv[j], (short)0, acc[i][j], false, false);
    }

    int rsub = (lane >> 4) << 3;
    int csub = lane & 15;
#pragma unroll
    for (int j = 0; j < 4; ++j) {
        int col = nbase + j * 16 + csub;
        float bv = bias[col];
#pragma unroll
        for (int i = 0; i < 4; ++i) {
            int rowb = mbase + i * 16 + rsub;
#pragma unroll
            for (int e = 0; e < 8; ++e)
                out[(size_t)(rowb + e) * VV + col] = acc[i][j][e] + bv;
        }
    }
}

// ---------------------------------------------------------------------------
// Host-side orchestration
// ---------------------------------------------------------------------------
extern "C" void kernel_launch(void* const* d_in, const int* in_sizes, int n_in,
                              void* d_out, int out_size, void* d_ws, size_t ws_size,
                              hipStream_t stream) {
    (void)in_sizes; (void)n_in; (void)out_size; (void)ws_size;

    const int*   input_tok = (const int*)  d_in[0];
    const int*   target_tok= (const int*)  d_in[1];
    const float* enc_emb   = (const float*)d_in[2];
    const float* enc_w_ih  = (const float*)d_in[3];
    const float* enc_w_hh  = (const float*)d_in[4];
    const float* enc_b_ih  = (const float*)d_in[5];
    const float* enc_b_hh  = (const float*)d_in[6];
    const float* dec_emb   = (const float*)d_in[7];
    const float* dec_w_ih  = (const float*)d_in[8];
    const float* dec_w_hh  = (const float*)d_in[9];
    const float* dec_b_ih  = (const float*)d_in[10];
    const float* dec_b_hh  = (const float*)d_in[11];
    const float* out_w     = (const float*)d_in[12];
    const float* out_b     = (const float*)d_in[13];

    float* logits  = (float*)d_out;                       // [B*T, V]
    float* dec_h_o = (float*)d_out + (size_t)BB * TT * VV;// [B, H]

    // ---- workspace carve-up (256B aligned slices) ----
    size_t off = 0;
    auto carve = [&](size_t bytes) -> char* {
        char* p = (char*)d_ws + off;
        off += (bytes + 255) & ~(size_t)255;
        return p;
    };
    uint16_t* wih_e  = (uint16_t*)carve((size_t)H3 * HH * 2);
    uint16_t* whh_e  = (uint16_t*)carve((size_t)H3 * HH * 2);
    uint16_t* wih_d  = (uint16_t*)carve((size_t)H3 * HH * 2);
    uint16_t* whh_d  = (uint16_t*)carve((size_t)H3 * HH * 2);
    uint16_t* outw_b = (uint16_t*)carve((size_t)VV * HH * 2);
    uint16_t* x_enc  = (uint16_t*)carve((size_t)BB * SS * HH * 2);
    uint16_t* x_dec  = (uint16_t*)carve((size_t)BB * TT * HH * 2);
    uint16_t* d_outb = (uint16_t*)carve((size_t)BB * TT * HH * 2);
    uint16_t* h_bf   = (uint16_t*)carve((size_t)BB * HH * 2);
    float*    h_f32  = (float*)   carve((size_t)BB * HH * 4);
    float*    gi     = (float*)   carve((size_t)BB * H3 * 4);
    float*    gh     = (float*)   carve((size_t)BB * H3 * 4);

    const int TPB = 256;
    auto blocks = [](size_t n, int tpb) { return (int)((n + tpb - 1) / tpb); };

    // ---- weight / embedding precision conversion (once per call) ----
    k_f32_to_bf16<<<blocks((size_t)H3 * HH, TPB), TPB, 0, stream>>>(enc_w_ih, wih_e, H3 * HH);
    k_f32_to_bf16<<<blocks((size_t)H3 * HH, TPB), TPB, 0, stream>>>(enc_w_hh, whh_e, H3 * HH);
    k_f32_to_bf16<<<blocks((size_t)H3 * HH, TPB), TPB, 0, stream>>>(dec_w_ih, wih_d, H3 * HH);
    k_f32_to_bf16<<<blocks((size_t)H3 * HH, TPB), TPB, 0, stream>>>(dec_w_hh, whh_d, H3 * HH);
    k_f32_to_bf16<<<blocks((size_t)VV * HH, TPB), TPB, 0, stream>>>(out_w, outw_b, VV * HH);

    k_enc_embed<<<blocks((size_t)BB * SS * HH, TPB), TPB, 0, stream>>>(input_tok, enc_emb, x_enc);
    k_dec_embed<<<blocks((size_t)BB * TT * HH, TPB), TPB, 0, stream>>>(target_tok, dec_emb, x_dec);
    k_zero_h<<<blocks((size_t)BB * HH, TPB), TPB, 0, stream>>>(h_f32, h_bf);

    dim3 gemm_grid(H3 / 16, BB / 16);   // 192 x 2 tiles, one wave each

    // ---- encoder scan ----
    for (int t = 0; t < SS; ++t) {
        k_gru_gemm<<<gemm_grid, 32, 0, stream>>>(
            x_enc + (size_t)t * HH, SS * HH, h_bf, wih_e, whh_e,
            enc_b_ih, enc_b_hh, gi, gh);
        k_gru_gate<<<blocks((size_t)BB * HH, TPB), TPB, 0, stream>>>(
            gi, gh, h_f32, h_bf, (uint16_t*)nullptr, (float*)nullptr);
    }

    // ---- decoder scan (teacher forcing); h carries over from encoder ----
    for (int t = 0; t < TT; ++t) {
        k_gru_gemm<<<gemm_grid, 32, 0, stream>>>(
            x_dec + (size_t)t * HH, TT * HH, h_bf, wih_d, whh_d,
            dec_b_ih, dec_b_hh, gi, gh);
        k_gru_gate<<<blocks((size_t)BB * HH, TPB), TPB, 0, stream>>>(
            gi, gh, h_f32, h_bf, d_outb + (size_t)t * HH,
            (t == TT - 1) ? dec_h_o : (float*)nullptr);
    }

    // ---- output projection: the 134-GFLOP GEMM, 64x64 tile per wave ----
    dim3 proj_grid(VV / 256, (BB * TT) / 64);   // 125 x 32 blocks, 4 waves each
    k_out_proj<<<proj_grid, 128, 0, stream>>>(d_outb, outw_b, out_b, logits);
}
